// Convolution_1451698946404
// MI455X (gfx1250) — matrix operand order repaired and checked by reading, not compiled, not run on hardware
//
#include <hip/hip_runtime.h>

typedef __attribute__((ext_vector_type(16))) _Float16 v16h;
typedef __attribute__((ext_vector_type(8)))  _Float16 v8h;
typedef __attribute__((ext_vector_type(8)))  float    v8f;

#define C_IN   16
#define C_OUT  16
#define RBASIS 8
#define NPTS   768
#define KDIM   (NPTS * RBASIS)   // 6144
#define KSTEPS (KDIM / 32)       // 192 WMMA K-steps
#define SPLITS 8                 // split-K factor -> 384 GEMM waves
#define STEPS_PER_SPLIT (KSTEPS / SPLITS)  // 24
#define EPS    1e-9f

// ---------------------------------------------------------------------------
// Kernel 1: Ht[i][b*8+r] = (1/sqrt(n_norm)) * sum_j W[r,i,j] * features[b,j]
// Transposed [C_OUT][KDIM] f16 so B-operand fragments are contiguous 32B.
// ---------------------------------------------------------------------------
__global__ void build_ht_kernel(const float* __restrict__ features,
                                const float* __restrict__ W,
                                const int*   __restrict__ n_norm,
                                _Float16*    __restrict__ Ht) {
    int t = blockIdx.x * blockDim.x + threadIdx.x;
    if (t >= C_OUT * KDIM) return;
    int i = t / KDIM;
    int k = t - i * KDIM;
    int b = k >> 3;
    int r = k & 7;
    float scale = rsqrtf((float)n_norm[0]);
    const float* w = W        + (r * C_OUT + i) * C_IN;
    const float* f = features + b * C_IN;
    float s = 0.0f;
#pragma unroll
    for (int j = 0; j < C_IN; ++j) s += w[j] * f[j];
    Ht[t] = (_Float16)(s * scale);
}

// ---------------------------------------------------------------------------
// Kernel 2: RBF[a][b*8+r] = exp(-gamma[r]*(d(a,b)-mu[r])^2), f16.
// One thread per (a,b) pair: full-chip parallelism for the 4.7M exps.
// Row-major [NPTS][KDIM] matches WMMA A-fragment addressing exactly.
// ---------------------------------------------------------------------------
__global__ void build_rbf_kernel(const float* __restrict__ geometry,
                                 const float* __restrict__ mu_p,
                                 const float* __restrict__ gam_p,
                                 _Float16*    __restrict__ RBF) {
    int t = blockIdx.x * blockDim.x + threadIdx.x;
    if (t >= NPTS * NPTS) return;
    int a = t / NPTS;
    int b = t - a * NPTS;
    const float dx = geometry[b * 3 + 0] - geometry[a * 3 + 0];
    const float dy = geometry[b * 3 + 1] - geometry[a * 3 + 1];
    const float dz = geometry[b * 3 + 2] - geometry[a * 3 + 2];
    const float d  = sqrtf(dx * dx + dy * dy + dz * dz + EPS);
    v8h v;
#pragma unroll
    for (int r = 0; r < RBASIS; ++r) {
        const float tr = d - mu_p[r];
        v[r] = (_Float16)__expf(-gam_p[r] * tr * tr);
    }
    *(v8h*)(RBF + (size_t)a * KDIM + b * RBASIS) = v;   // 16B aligned store
}

// ---------------------------------------------------------------------------
// A-fragment loader: 16-bit A 16x32, contiguous k order (not ISA table order).
// lane(m,hl): halves 0..15 -> K = hl*16+e (contiguous, matches B).
// Two adjacent 16B loads -> compiler merges to paired global_load_b128.
// ---------------------------------------------------------------------------
__device__ __forceinline__ v16h load_a_frag(const _Float16* __restrict__ p) {
    const v8h lo = *(const v8h*)(p);
    const v8h hi = *(const v8h*)(p + 8);
    return __builtin_shufflevector(lo, hi,
        0, 1, 2, 3, 4, 5, 6, 7, 8, 9, 10, 11, 12, 13, 14, 15);
}

// ---------------------------------------------------------------------------
// Kernel 3: split-K WMMA GEMM, software-pipelined (loads for step s+1 issue
// before the WMMA of step s; unconditional speculative prefetch 8 steps out).
// grid = (12, SPLITS), block = 128 (4 waves); one 16-row tile per wave.
// ---------------------------------------------------------------------------
__global__ void __launch_bounds__(128)
gemm_wmma_kernel(const _Float16* __restrict__ RBF,
                 const _Float16* __restrict__ Ht,
                 float*          __restrict__ part) {
    const int wave  = threadIdx.x >> 5;
    const int lane  = threadIdx.x & 31;
    const int tile  = blockIdx.x * 4 + wave;   // 0..47
    const int split = blockIdx.y;              // 0..SPLITS-1
    const int hl    = lane >> 4;
    const int m     = lane & 15;

    const size_t kbase = (size_t)split * STEPS_PER_SPLIT * 32;
    const _Float16* ap = RBF + (size_t)(tile * 16 + m) * KDIM + kbase + hl * 16;
    const _Float16* bp = Ht  + (size_t)m * KDIM               + kbase + hl * 16;

    v8f acc = {};
    v16h A = load_a_frag(ap);
    v16h B = *(const v16h*)(bp);

#pragma unroll 4
    for (int s = 0; s < STEPS_PER_SPLIT - 1; ++s) {
        // speculative prefetch (silently dropped if OOB) — no branch
        __builtin_prefetch(ap + (s + 8) * 32, 0, 1);
        __builtin_prefetch(bp + (s + 8) * 32, 0, 1);
        const v16h An = load_a_frag(ap + (s + 1) * 32);
        const v16h Bn = *(const v16h*)(bp + (s + 1) * 32);
        acc = __builtin_amdgcn_wmma_f32_16x16x32_f16(
            false, A, false, B, (short)0, acc, false, false);
        A = An;
        B = Bn;
    }
    acc = __builtin_amdgcn_wmma_f32_16x16x32_f16(
        false, A, false, B, (short)0, acc, false, false);

    // D layout: VGPR v -> row = 8*hl + v, col = m
    float* pbase = part + (size_t)split * NPTS * C_OUT;
#pragma unroll
    for (int v = 0; v < 8; ++v) {
        const int row = tile * 16 + hl * 8 + v;
        pbase[row * C_OUT + m] = acc[v];
    }
}

// ---------------------------------------------------------------------------
// Kernel 4: deterministic reduction of split-K partials.
// ---------------------------------------------------------------------------
__global__ void reduce_kernel(const float* __restrict__ part,
                              float*       __restrict__ out) {
    int t = blockIdx.x * blockDim.x + threadIdx.x;
    if (t >= NPTS * C_OUT) return;
    float s = 0.0f;
#pragma unroll
    for (int sp = 0; sp < SPLITS; ++sp)
        s += part[(size_t)sp * NPTS * C_OUT + t];
    out[t] = s;
}

// ---------------------------------------------------------------------------
extern "C" void kernel_launch(void* const* d_in, const int* in_sizes, int n_in,
                              void* d_out, int out_size, void* d_ws, size_t ws_size,
                              hipStream_t stream) {
    const float* features = (const float*)d_in[0];   // [1,768,16]
    const float* geometry = (const float*)d_in[1];   // [1,768,3]
    const float* W        = (const float*)d_in[2];   // [8,16,16]
    const float* mu       = (const float*)d_in[3];   // [8]
    const float* gamma_v  = (const float*)d_in[4];   // [8]
    const int*   n_norm   = (const int*)d_in[5];     // scalar

    float* out = (float*)d_out;                      // [1,768,16] f32

    // Workspace layout (256B-aligned offsets):
    //   Ht   : [16][6144] f16          = 196608 B
    //   RBF  : [768][6144] f16         = 9437184 B
    //   part : [SPLITS][768][16] f32   = 393216 B
    char* ws = (char*)d_ws;
    _Float16* Ht   = (_Float16*)(ws);
    _Float16* RBF  = (_Float16*)(ws + 196608);
    float*    part = (float*)   (ws + 196608 + 9437184);

    {   // Stage 1: Ht = W-contracted features (f16, transposed)
        const int total = C_OUT * KDIM;              // 98304
        build_ht_kernel<<<(total + 127) / 128, 128, 0, stream>>>(
            features, W, n_norm, Ht);
    }
    {   // Stage 2: materialize RBF at full-chip parallelism
        const int total = NPTS * NPTS;               // 589824
        build_rbf_kernel<<<(total + 255) / 256, 256, 0, stream>>>(
            geometry, mu, gamma_v, RBF);
    }
    // Stage 3: split-K WMMA GEMM into partials (deterministic, 384 waves)
    gemm_wmma_kernel<<<dim3(12, SPLITS), 128, 0, stream>>>(RBF, Ht, part);
    // Stage 4: reduce partials
    reduce_kernel<<<(NPTS * C_OUT + 255) / 256, 256, 0, stream>>>(part, out);
}
